// PiNet2P5Prod_36017595744837
// MI455X (gfx1250) — matrix operand order repaired and checked by reading, not compiled, not run on hardware
//
#include <hip/hip_runtime.h>

// ---------------------------------------------------------------------------
// PiNet2-P5 forward for MI455X (gfx1250, wave32, WMMA).
// Per-pair dense stacks run on v_wmma_f32_16x16x32_f16, 16 pairs per wave.
// All packed weights K-padded to 32 -> branch-free B fragments; B fragments
// hoisted into registers across inner loops. Segment sums via fp32 atomics.
// tanh via native v_tanh_f32.
// ---------------------------------------------------------------------------

#define NATOMS 8192
#define NPAIRS 262144
#define PI_F 3.14159265358979f
#define RC_F 4.0f

#define WAVES_PER_BLOCK 4
#define TILES_PER_WAVE 4

// packed f16 weight blob offsets (halves); every matrix is [n][32] (K padded)
#define OFF_PI1A 0      // 16 n
#define OFF_PI1B 512    // 64 n
#define OFF_II1A 2560   // 16 n
#define OFF_II1B 3072   // 64 n
#define OFF_II3A 5120   // 16 n
#define OFF_II3B 5632
#define OFF_II5A 6144
#define OFF_II5B 6656
#define OFF_PI3I 7168
#define OFF_PI3J 7680
#define OFF_PI5I 8192
#define OFF_PI5J 8704
#define WBLOB_HALVES 9216
#define BBLOB_FLOATS 80

typedef __attribute__((ext_vector_type(16))) _Float16 v16h;
typedef __attribute__((ext_vector_type(8)))  _Float16 v8h;
typedef __attribute__((ext_vector_type(8)))  float    v8f;

union U16 { v16h v; struct { v8h lo; v8h hi; } s; };

__device__ __forceinline__ v8h zero8h() {
  v8h z;
#pragma unroll
  for (int i = 0; i < 8; ++i) z[i] = (_Float16)0.f;
  return z;
}
__device__ __forceinline__ v8f zero8f() {
  v8f z;
#pragma unroll
  for (int i = 0; i < 8; ++i) z[i] = 0.f;
  return z;
}

__device__ __forceinline__ v8f wmma16(v16h a, v16h b, v8f c) {
  // emits v_wmma_f32_16x16x32_f16
  return __builtin_amdgcn_wmma_f32_16x16x32_f16(false, a, false, b, (short)0, c,
                                                false, false);
}

__device__ __forceinline__ float fast_tanh(float x) {
#if __has_builtin(__builtin_amdgcn_tanhf)
  return __builtin_amdgcn_tanhf(x);   // v_tanh_f32, 1 TRANS op
#else
  x = fminf(10.f, fmaxf(-10.f, x));
  float e = __builtin_amdgcn_exp2f(x * 2.8853900817779268f);
  return (e - 1.f) * __builtin_amdgcn_rcpf(e + 1.f);
#endif
}

// DS ordering barrier within a wave (stores visible to subsequent loads)
#define LDSB() asm volatile("s_wait_dscnt 0x0" ::: "memory")

// ---- WMMA fragment builders ------------------------------------------------
// A frag (16xK rows = pairs), lane: m = lane&15, hi = lane>>4.
//   lo 8 halves = K [8*hi, 8*hi+8), hi 8 halves = K [16+8*hi, 16+8*hi+8)
__device__ __forceinline__ v16h a_gather32(const _Float16* ri, const _Float16* rj,
                                           int hi) {
  U16 u;
  u.s.lo = *(const v8h*)(ri + 8 * hi);
  u.s.hi = *(const v8h*)(rj + 8 * hi);
  return u.v;
}
__device__ __forceinline__ v16h a_gather16(const _Float16* ri, int hi) {
  U16 u;
  u.s.lo = *(const v8h*)(ri + 8 * hi);
  u.s.hi = zero8h();
  return u.v;
}
// LDS A-tile [16][32] halves, cols 16..31 pre-zeroed
__device__ __forceinline__ v16h a_lds(const _Float16* aBuf, int m, int hi) {
  U16 u;
  u.s.lo = *(const v8h*)(aBuf + m * 32 + 8 * hi);
  u.s.hi = *(const v8h*)(aBuf + m * 32 + 16 + 8 * hi);
  return u.v;
}
// B frag: lane: n = lane&15, hi; lo 8 halves = K [8*hi, 8*hi+8), hi 8 halves = K [16+8*hi, 16+8*hi+8).
// All matrices K-padded to 32 -> branch-free.
__device__ __forceinline__ v16h b_frag(const _Float16* wt, int n, int hi) {
  U16 u;
  const _Float16* p = wt + n * 32 + 8 * hi;
  u.s.lo = *(const v8h*)p;
  u.s.hi = *(const v8h*)(p + 16);
  return u.v;
}
// C frag (m = 8*hi + r, col n) -> LDS A-tile (f16)
__device__ __forceinline__ void dump_c(_Float16* aBuf, const v8f& c, int n, int hi) {
#pragma unroll
  for (int r = 0; r < 8; ++r) aBuf[(8 * hi + r) * 32 + n] = (_Float16)c[r];
  LDSB();
}

// ---- per-wave LDS arena ----------------------------------------------------
struct Arena {
  alignas(16) _Float16 aBuf[16 * 32];   // generic A-tile
  alignas(16) _Float16 aBuf2[16 * 32];  // i1 A-tile
  float cBuf[16 * 16];                  // basis-contraction scratch
  float i2Buf[16 * 16];                 // i1_2
  float i3Buf[16 * 16];                 // i1_3
  float i4Buf[16 * 16];                 // i1_4
  float sdBuf[16 * 48];                 // scaled_diff [pair][x][c]
  int idxI[16];
  int idxJ[16];
  float nd[48];                         // ndiff [x][pair]
  float basis[64];                      // [pair][4]
};

// ---------------------------------------------------------------------------
__global__ void init_kernel(const int* __restrict__ elems, float* p1, float* p3,
                            float* p5, float* p1n, float* p3n, float* p5n,
                            float* out) {
  int a = blockIdx.x * blockDim.x + threadIdx.x;
  if (a >= NATOMS) return;
  int e = elems[a];
  for (int c = 0; c < 16; ++c) {
    p1[(size_t)a * 16 + c] = (c == e) ? 1.f : 0.f;
    p1n[(size_t)a * 16 + c] = 0.f;
  }
  for (int i = 0; i < 48; ++i) {
    p3[(size_t)a * 48 + i] = 0.f;
    p3n[(size_t)a * 48 + i] = 0.f;
  }
  for (int i = 0; i < 144; ++i) {
    p5[(size_t)a * 144 + i] = 0.f;
    p5n[(size_t)a * 144 + i] = 0.f;
  }
  out[a] = 0.f;
}

// ---------------------------------------------------------------------------
// Pack per-depth pair weights to fp16 B-fragment layout [n][32], K zero-padded.
__global__ void pack_kernel(const float* pi1aw, const float* pi1ab,
                            const float* pi1bw, const float* pi1bb,
                            const float* ii1a, const float* ii1b,
                            const float* ii3a, const float* ii3b,
                            const float* ii5a, const float* ii5b,
                            const float* pi3i, const float* pi3j,
                            const float* pi5i, const float* pi5j,
                            _Float16* wb, float* bb, int pcx, int first) {
  int g = blockIdx.x * blockDim.x + threadIdx.x;
  if (g < WBLOB_HALVES) {
    float v = 0.f;
    if (g < 512) {  // pi1 layer A: K=32; d0: concat(4)+pad -> k<4 and 16<=k<20
      int nn = g / 32, k = g % 32;
      if (first) {
        if (k < 4) v = pi1aw[k * 16 + nn];
        else if (k >= 16 && k < 20) v = pi1aw[(k - 12) * 16 + nn];
      } else {
        v = pi1aw[k * 16 + nn];
      }
    } else if (g < 2560) {  // pi1 layer B: w(16,64)
      int q = g - 512, nn = q / 32, k = q % 32;
      if (k < 16) v = pi1bw[k * 64 + nn];
    } else if (g < 3072) {  // ii1a (16,16)
      int q = g - 2560, nn = q / 32, k = q % 32;
      if (k < 16) v = ii1a[k * 16 + nn];
    } else if (g < 5120) {  // ii1b (16,64)
      int q = g - 3072, nn = q / 32, k = q % 32;
      if (k < 16) v = ii1b[k * 64 + nn];
    } else if (g < 5632) {  // ii3a (pcx,16)
      int q = g - 5120, nn = q / 32, k = q % 32;
      if (k < pcx) v = ii3a[k * 16 + nn];
    } else if (g < 6144) {  // ii3b (16,16)
      int q = g - 5632, nn = q / 32, k = q % 32;
      if (k < 16) v = ii3b[k * 16 + nn];
    } else if (g < 6656) {  // ii5a (pcx,16)
      int q = g - 6144, nn = q / 32, k = q % 32;
      if (k < pcx) v = ii5a[k * 16 + nn];
    } else if (g < 7168) {  // ii5b (16,16)
      int q = g - 6656, nn = q / 32, k = q % 32;
      if (k < 16) v = ii5b[k * 16 + nn];
    } else if (g < 7680) {  // pi3 wi (pcx,pcx)
      int q = g - 7168, nn = q / 32, k = q % 32;
      if (k < pcx && nn < pcx) v = pi3i[k * pcx + nn];
    } else if (g < 8192) {
      int q = g - 7680, nn = q / 32, k = q % 32;
      if (k < pcx && nn < pcx) v = pi3j[k * pcx + nn];
    } else if (g < 8704) {
      int q = g - 8192, nn = q / 32, k = q % 32;
      if (k < pcx && nn < pcx) v = pi5i[k * pcx + nn];
    } else {
      int q = g - 8704, nn = q / 32, k = q % 32;
      if (k < pcx && nn < pcx) v = pi5j[k * pcx + nn];
    }
    wb[g] = (_Float16)v;
  } else if (g < WBLOB_HALVES + BBLOB_FLOATS) {
    int q = g - WBLOB_HALVES;
    bb[q] = (q < 16) ? pi1ab[q] : pi1bb[q - 16];
  }
}

// ---------------------------------------------------------------------------
// Per-atom property layers: h1 = pp1(p1), h3 = pp3(p3), h5 = pp5(p5) -> fp16.
__global__ void prep_kernel(const float* __restrict__ p1,
                            const float* __restrict__ p3,
                            const float* __restrict__ p5, _Float16* h1h,
                            _Float16* h3h, _Float16* h5h, const float* pp1w1,
                            const float* pp1b1, const float* pp1w2,
                            const float* pp1b2, const float* pp3w1,
                            const float* pp3w2, const float* pp5w1,
                            const float* pp5b1, const float* pp5w2,
                            const float* pp5b2, int first) {
  int gid = blockIdx.x * blockDim.x + threadIdx.x;
  if (gid >= NATOMS * 13) return;
  int a = gid / 13, u = gid % 13;
  const float* src;
  _Float16* dst;
  if (u == 0) { src = p1 + (size_t)a * 16; dst = h1h + (size_t)a * 16; }
  else if (u < 4) { src = p3 + (size_t)a * 48 + (u - 1) * 16; dst = h3h + (size_t)a * 48 + (u - 1) * 16; }
  else { src = p5 + (size_t)a * 144 + (u - 4) * 16; dst = h5h + (size_t)a * 144 + (u - 4) * 16; }
  if (first) {  // first block: pp* are identity
    for (int c = 0; c < 16; ++c) dst[c] = (_Float16)src[c];
    return;
  }
  const float *w1, *b1, *w2, *b2;
  int act;
  if (u == 0) { w1 = pp1w1; b1 = pp1b1; w2 = pp1w2; b2 = pp1b2; act = 1; }
  else if (u < 4) { w1 = pp3w1; b1 = nullptr; w2 = pp3w2; b2 = nullptr; act = 0; }
  else { w1 = pp5w1; b1 = pp5b1; w2 = pp5w2; b2 = pp5b2; act = 1; }
  float in[16], t1[16];
  for (int k = 0; k < 16; ++k) in[k] = src[k];
  for (int o = 0; o < 16; ++o) {
    float s = b1 ? b1[o] : 0.f;
    for (int k = 0; k < 16; ++k) s = fmaf(in[k], w1[k * 16 + o], s);
    t1[o] = act ? fast_tanh(s) : s;
  }
  for (int o = 0; o < 16; ++o) {
    float s = b2 ? b2[o] : 0.f;
    for (int k = 0; k < 16; ++k) s = fmaf(t1[k], w2[k * 16 + o], s);
    dst[o] = (_Float16)(act ? fast_tanh(s) : s);
  }
}

// ---------------------------------------------------------------------------
// The WMMA pair kernel: one wave = 16 pairs, 58 WMMAs per tile.
__global__ __launch_bounds__(128) void pair_kernel(
    const int* __restrict__ ind2, const float* __restrict__ diff,
    const _Float16* __restrict__ h1h, const _Float16* __restrict__ h3h,
    const _Float16* __restrict__ h5h, const _Float16* __restrict__ wblob,
    const float* __restrict__ bblob, float* __restrict__ p1n,
    float* __restrict__ p3n, float* __restrict__ p5n) {
  __shared__ alignas(16) _Float16 sWt[WBLOB_HALVES];
  __shared__ float sB[BBLOB_FLOATS];
  __shared__ Arena ar[WAVES_PER_BLOCK];

  const int tid = threadIdx.x;
  for (int i = tid; i < WBLOB_HALVES / 8; i += 128)
    ((uint4*)sWt)[i] = ((const uint4*)wblob)[i];
  for (int i = tid; i < BBLOB_FLOATS; i += 128) sB[i] = bblob[i];
  __syncthreads();

  const int wv = tid >> 5, lane = tid & 31;
  const int nl = lane & 15, hi = lane >> 4;
  Arena& A = ar[wv];
  for (int i = lane; i < 256; i += 32) {  // zero K=16..31 columns once
    A.aBuf[(i >> 4) * 32 + 16 + (i & 15)] = (_Float16)0.f;
    A.aBuf2[(i >> 4) * 32 + 16 + (i & 15)] = (_Float16)0.f;
  }
  LDSB();

  const float bA = sB[nl];
  float bB[4];
#pragma unroll
  for (int t = 0; t < 4; ++t) bB[t] = sB[16 + 16 * t + nl];

  const int gwave = blockIdx.x * WAVES_PER_BLOCK + wv;
  for (int tt = 0; tt < TILES_PER_WAVE; ++tt) {
    const int pbase = (gwave * TILES_PER_WAVE + tt) * 16;
    if (lane < 16) {  // per-pair scalars
      const int p = pbase + lane;
      const int ii = ind2[2 * p], jj = ind2[2 * p + 1];
      const float dx = diff[3 * p], dy = diff[3 * p + 1], dz = diff[3 * p + 2];
      const float r2 = dx * dx + dy * dy + dz * dz;
      const float inv = rsqrtf(r2);
      const float dist = r2 * inv;
      const float fc = 0.5f * (__cosf(PI_F * dist * (1.0f / RC_F)) + 1.0f);
      A.idxI[lane] = ii;
      A.idxJ[lane] = jj;
      A.nd[0 * 16 + lane] = dx * inv;
      A.nd[1 * 16 + lane] = dy * inv;
      A.nd[2 * 16 + lane] = dz * inv;
      const float f2 = fc * fc;
      A.basis[lane * 4 + 0] = fc;
      A.basis[lane * 4 + 1] = f2;
      A.basis[lane * 4 + 2] = f2 * fc;
      A.basis[lane * 4 + 3] = f2 * f2;
    }
    LDSB();
    const int gI = A.idxI[nl], gJ = A.idxJ[nl];
    int rI[8];
    float bas[8];
#pragma unroll
    for (int r = 0; r < 8; ++r) {
      rI[r] = A.idxI[8 * hi + r];
      bas[r] = A.basis[(8 * hi + r) * 4 + (nl & 3)];
    }

    // ---- pi1 layer A: inter(16x32) @ W -> tanh ----
    v8f c = wmma16(a_gather32(h1h + (size_t)gI * 16, h1h + (size_t)gJ * 16, hi),
                   b_frag(sWt + OFF_PI1A, nl, hi), zero8f());
#pragma unroll
    for (int r = 0; r < 8; ++r) c[r] = fast_tanh(c[r] + bA);
    dump_c(A.aBuf, c, nl, hi);

    // ---- pi1 layer B (x4) fused with basis contraction -> i1 tile ----
    {
      const v16h aI = a_lds(A.aBuf, nl, hi);
      v16h Bb[4];
#pragma unroll
      for (int t4 = 0; t4 < 4; ++t4)
        Bb[t4] = b_frag(sWt + OFF_PI1B + t4 * 512, nl, hi);
      for (int t4 = 0; t4 < 4; ++t4) {
        v8f ct = wmma16(aI, Bb[t4], zero8f());
#pragma unroll
        for (int r = 0; r < 8; ++r) {
          float v = fast_tanh(ct[r] + bB[t4]);
          A.cBuf[(8 * hi + r) * 16 + nl] = v * bas[r];
        }
        LDSB();
        if (lane < 16) {
#pragma unroll
          for (int cl = 0; cl < 4; ++cl) {
            float s = A.cBuf[lane * 16 + 4 * cl + 0] +
                      A.cBuf[lane * 16 + 4 * cl + 1] +
                      A.cBuf[lane * 16 + 4 * cl + 2] +
                      A.cBuf[lane * 16 + 4 * cl + 3];
            A.aBuf2[lane * 32 + 4 * t4 + cl] = (_Float16)s;
          }
        }
        LDSB();
      }
    }

    // ---- ii1: dense->tanh, dense(64)->tanh; split i1_1..i1_4 ----
    c = wmma16(a_lds(A.aBuf2, nl, hi), b_frag(sWt + OFF_II1A, nl, hi), zero8f());
#pragma unroll
    for (int r = 0; r < 8; ++r) c[r] = fast_tanh(c[r]);
    dump_c(A.aBuf, c, nl, hi);
    {
      const v16h a2 = a_lds(A.aBuf, nl, hi);
      v16h Bb[4];
#pragma unroll
      for (int t4 = 0; t4 < 4; ++t4)
        Bb[t4] = b_frag(sWt + OFF_II1B + t4 * 512, nl, hi);
      for (int t4 = 0; t4 < 4; ++t4) {
        v8f ct = wmma16(a2, Bb[t4], zero8f());
#pragma unroll
        for (int r = 0; r < 8; ++r) ct[r] = fast_tanh(ct[r]);
        if (t4 == 0) {  // i1_1 -> p1_new segment sum
#pragma unroll
          for (int r = 0; r < 8; ++r)
            atomicAdd(p1n + (size_t)rI[r] * 16 + nl, ct[r]);
        } else {
          float* dst = (t4 == 1) ? A.i2Buf : (t4 == 2) ? A.i3Buf : A.i4Buf;
#pragma unroll
          for (int r = 0; r < 8; ++r) dst[(8 * hi + r) * 16 + nl] = ct[r];
        }
      }
    }
    LDSB();

    // ---- scaled_diff[pair][x][c] = ndiff * i1_2 ----
    if (lane < 16) {
      for (int x = 0; x < 3; ++x) {
        const float nv = A.nd[x * 16 + lane];
#pragma unroll
        for (int cc = 0; cc < 16; ++cc)
          A.sdBuf[lane * 48 + x * 16 + cc] = nv * A.i2Buf[lane * 16 + cc];
      }
    }
    LDSB();

    // ---- p3 path: (pi3 gather-matmul x2) -> ii3 -> scale+add -> scatter ----
    {
      const v16h Bp3i = b_frag(sWt + OFF_PI3I, nl, hi);
      const v16h Bp3j = b_frag(sWt + OFF_PI3J, nl, hi);
      const v16h Bi3a = b_frag(sWt + OFF_II3A, nl, hi);
      const v16h Bi3b = b_frag(sWt + OFF_II3B, nl, hi);
      for (int x = 0; x < 3; ++x) {
        c = wmma16(a_gather16(h3h + (size_t)gJ * 48 + x * 16, hi), Bp3j,
                   zero8f());
        c = wmma16(a_gather16(h3h + (size_t)gI * 48 + x * 16, hi), Bp3i, c);
        dump_c(A.aBuf, c, nl, hi);
        c = wmma16(a_lds(A.aBuf, nl, hi), Bi3a, zero8f());
#pragma unroll
        for (int r = 0; r < 8; ++r) c[r] = fast_tanh(c[r]);
        dump_c(A.aBuf, c, nl, hi);
        c = wmma16(a_lds(A.aBuf, nl, hi), Bi3b, zero8f());
#pragma unroll
        for (int r = 0; r < 8; ++r) {
          const int mr = 8 * hi + r;
          float v = fast_tanh(c[r]) * A.i3Buf[mr * 16 + nl] +
                    A.sdBuf[mr * 48 + x * 16 + nl];
          atomicAdd(p3n + (size_t)rI[r] * 48 + x * 16 + nl, v);
        }
      }
    }

    // ---- p5 path: 9 spatial tiles ----
    {
      const v16h Bp5i = b_frag(sWt + OFF_PI5I, nl, hi);
      const v16h Bp5j = b_frag(sWt + OFF_PI5J, nl, hi);
      const v16h Bi5a = b_frag(sWt + OFF_II5A, nl, hi);
      const v16h Bi5b = b_frag(sWt + OFF_II5B, nl, hi);
      for (int xy = 0; xy < 9; ++xy) {
        const int x = xy / 3, y = xy % 3;
        c = wmma16(a_gather16(h5h + (size_t)gJ * 144 + xy * 16, hi), Bp5j,
                   zero8f());
        c = wmma16(a_gather16(h5h + (size_t)gI * 144 + xy * 16, hi), Bp5i, c);
        dump_c(A.aBuf, c, nl, hi);
        c = wmma16(a_lds(A.aBuf, nl, hi), Bi5a, zero8f());
#pragma unroll
        for (int r = 0; r < 8; ++r) c[r] = fast_tanh(c[r]);
        dump_c(A.aBuf, c, nl, hi);
        c = wmma16(a_lds(A.aBuf, nl, hi), Bi5b, zero8f());
#pragma unroll
        for (int r = 0; r < 8; ++r) {
          const int mr = 8 * hi + r;
          const float sx = A.sdBuf[mr * 48 + x * 16 + nl];
          const float sy = A.sdBuf[mr * 48 + y * 16 + nl];
          float v = fast_tanh(c[r]) * A.i4Buf[mr * 16 + nl] + sx * sy;
          atomicAdd(p5n + (size_t)rI[r] * 144 + xy * 16 + nl, v);
        }
      }
    }
    LDSB();
  }
}

// ---------------------------------------------------------------------------
// Per-atom: p3.p5.p3 einsum, output head, residuals, clear accumulators.
__global__ void post_kernel(float* p1, float* p3, float* p5, float* p1n,
                            float* p3n, float* p5n, const float* ffw1,
                            const float* ffb1, const float* ffw2,
                            const float* ffb2, const float* outw,
                            const float* res1w, const float* res3w,
                            const float* res5w, float* out, int first) {
  int a = blockIdx.x * blockDim.x + threadIdx.x;
  if (a >= NATOMS) return;
  float q3[48];
  for (int i = 0; i < 48; ++i) q3[i] = p3n[(size_t)a * 48 + i];
  float t[16];
  for (int cc = 0; cc < 16; ++cc) {
    float s = p1n[(size_t)a * 16 + cc];
    for (int x = 0; x < 3; ++x)
      for (int y = 0; y < 3; ++y)
        s += q3[x * 16 + cc] * p5n[(size_t)a * 144 + (3 * x + y) * 16 + cc] *
             q3[y * 16 + cc];
    t[cc] = s;
  }
  float y1[16];
  for (int o = 0; o < 16; ++o) {
    float s = ffb1[o];
    for (int k = 0; k < 16; ++k) s = fmaf(t[k], ffw1[k * 16 + o], s);
    y1[o] = s;
  }
  float acc = 0.f;
  for (int o = 0; o < 16; ++o) {
    float s = ffb2[o];
    for (int k = 0; k < 16; ++k) s = fmaf(y1[k], ffw2[k * 16 + o], s);
    acc = fmaf(s, outw[o], acc);
  }
  out[a] += acc;
  if (first) {
    float r1[16];
    for (int cc = 0; cc < 16; ++cc) {
      float s = 0.f;
      for (int e = 0; e < 4; ++e)
        s = fmaf(p1[(size_t)a * 16 + e], res1w[e * 16 + cc], s);
      r1[cc] = s;
    }
    for (int cc = 0; cc < 16; ++cc) p1[(size_t)a * 16 + cc] = r1[cc] + t[cc];
    for (int x = 0; x < 3; ++x) {
      float old0 = p3[(size_t)a * 48 + x * 16 + 0];
      for (int cc = 0; cc < 16; ++cc)
        p3[(size_t)a * 48 + x * 16 + cc] =
            old0 * res3w[cc] + q3[x * 16 + cc] * t[cc];
    }
    for (int xy = 0; xy < 9; ++xy) {
      float old0 = p5[(size_t)a * 144 + xy * 16 + 0];
      for (int cc = 0; cc < 16; ++cc)
        p5[(size_t)a * 144 + xy * 16 + cc] =
            old0 * res5w[cc] + p5n[(size_t)a * 144 + xy * 16 + cc] * t[cc];
    }
  } else {
    for (int cc = 0; cc < 16; ++cc) p1[(size_t)a * 16 + cc] += t[cc];
    for (int x = 0; x < 3; ++x)
      for (int cc = 0; cc < 16; ++cc)
        p3[(size_t)a * 48 + x * 16 + cc] += q3[x * 16 + cc] * t[cc];
    for (int xy = 0; xy < 9; ++xy)
      for (int cc = 0; cc < 16; ++cc)
        p5[(size_t)a * 144 + xy * 16 + cc] +=
            p5n[(size_t)a * 144 + xy * 16 + cc] * t[cc];
  }
  for (int cc = 0; cc < 16; ++cc) p1n[(size_t)a * 16 + cc] = 0.f;
  for (int i = 0; i < 48; ++i) p3n[(size_t)a * 48 + i] = 0.f;
  for (int i = 0; i < 144; ++i) p5n[(size_t)a * 144 + i] = 0.f;
}

// ---------------------------------------------------------------------------
extern "C" void kernel_launch(void* const* d_in, const int* in_sizes, int n_in,
                              void* d_out, int out_size, void* d_ws,
                              size_t ws_size, hipStream_t stream) {
  (void)in_sizes; (void)n_in; (void)out_size; (void)ws_size;
  // setup_inputs order: ind_1, ind_2, elems_idx, diff, params(tree-flattened,
  // dict keys alphabetical). ind_1 (d_in[0]) unused by the reference math.
  const int* ind2 = (const int*)d_in[1];
  const int* elems = (const int*)d_in[2];
  const float* diff = (const float*)d_in[3];
  auto F = [&](int i) -> const float* { return (const float*)d_in[i]; };

  char* w = (char*)d_ws;
  auto carve = [&](size_t bytes) -> char* {
    char* p = w;
    w += (bytes + 255) & ~(size_t)255;
    return p;
  };
  float* p1 = (float*)carve(NATOMS * 16 * sizeof(float));
  float* p3 = (float*)carve(NATOMS * 48 * sizeof(float));
  float* p5 = (float*)carve(NATOMS * 144 * sizeof(float));
  float* p1n = (float*)carve(NATOMS * 16 * sizeof(float));
  float* p3n = (float*)carve(NATOMS * 48 * sizeof(float));
  float* p5n = (float*)carve(NATOMS * 144 * sizeof(float));
  _Float16* h1h = (_Float16*)carve(NATOMS * 16 * sizeof(_Float16));
  _Float16* h3h = (_Float16*)carve(NATOMS * 48 * sizeof(_Float16));
  _Float16* h5h = (_Float16*)carve(NATOMS * 144 * sizeof(_Float16));
  _Float16* wblob[4];
  float* bblob[4];
  for (int d = 0; d < 4; ++d) {
    wblob[d] = (_Float16*)carve(WBLOB_HALVES * sizeof(_Float16));
    bblob[d] = (float*)carve(BBLOB_FLOATS * sizeof(float));
  }
  float* out = (float*)d_out;

  init_kernel<<<dim3(NATOMS / 256), dim3(256), 0, stream>>>(elems, p1, p3, p5,
                                                            p1n, p3n, p5n, out);

  for (int d = 0; d < 4; ++d) {
    const int first = (d == 0);
    // param leaf indices: blocks start at 4; block0 has 14 leaves (pp* empty),
    // blocks 1-3 have 24; out heads at 90+5d; res at 110/111/112.
    const int base = first ? 4 : (18 + 24 * (d - 1));
    const float* ii1a = F(base + 0);
    const float* ii1b = F(base + 1);
    const float* ii3a = F(base + 2);
    const float* ii3b = F(base + 3);
    const float* ii5a = F(base + 4);
    const float* ii5b = F(base + 5);
    const float* pi1ab = F(base + 6);
    const float* pi1aw = F(base + 7);
    const float* pi1bb = F(base + 8);
    const float* pi1bw = F(base + 9);
    const float* pi3i = F(base + 10);
    const float* pi3j = F(base + 11);
    const float* pi5i = F(base + 12);
    const float* pi5j = F(base + 13);
    const float *pp1b1 = nullptr, *pp1w1 = nullptr, *pp1b2 = nullptr,
                *pp1w2 = nullptr, *pp3w1 = nullptr, *pp3w2 = nullptr,
                *pp5b1 = nullptr, *pp5w1 = nullptr, *pp5b2 = nullptr,
                *pp5w2 = nullptr;
    if (!first) {
      pp1b1 = F(base + 14); pp1w1 = F(base + 15);
      pp1b2 = F(base + 16); pp1w2 = F(base + 17);
      pp3w1 = F(base + 18); pp3w2 = F(base + 19);
      pp5b1 = F(base + 20); pp5w1 = F(base + 21);
      pp5b2 = F(base + 22); pp5w2 = F(base + 23);
    }
    const int ob = 90 + 5 * d;
    const float* ffb1 = F(ob + 0);
    const float* ffw1 = F(ob + 1);
    const float* ffb2 = F(ob + 2);
    const float* ffw2 = F(ob + 3);
    const float* outw = F(ob + 4);
    const float* res1w = F(110);
    const float* res3w = F(111);
    const float* res5w = F(112);

    pack_kernel<<<dim3((WBLOB_HALVES + BBLOB_FLOATS + 255) / 256, 1, 1),
                  dim3(256), 0, stream>>>(
        pi1aw, pi1ab, pi1bw, pi1bb, ii1a, ii1b, ii3a, ii3b, ii5a, ii5b, pi3i,
        pi3j, pi5i, pi5j, wblob[d], bblob[d], first ? 1 : 16, first);
    prep_kernel<<<dim3((NATOMS * 13) / 256), dim3(256), 0, stream>>>(
        p1, p3, p5, h1h, h3h, h5h, pp1w1, pp1b1, pp1w2, pp1b2, pp3w1, pp3w2,
        pp5w1, pp5b1, pp5w2, pp5b2, first);
    pair_kernel<<<dim3(NPAIRS / (16 * WAVES_PER_BLOCK * TILES_PER_WAVE)),
                  dim3(32 * WAVES_PER_BLOCK), 0, stream>>>(
        ind2, diff, h1h, h3h, h5h, wblob[d], bblob[d], p1n, p3n, p5n);
    post_kernel<<<dim3(NATOMS / 256), dim3(256), 0, stream>>>(
        p1, p3, p5, p1n, p3n, p5n, ffw1, ffb1, ffw2, ffb2, outw, res1w, res3w,
        res5w, out, first);
  }
}
